// NestedTensorBlock_10797547782322
// MI455X (gfx1250) — compile-verified, matrix operand order checked by reading, not run on hardware
//
#include <hip/hip_runtime.h>

typedef __attribute__((ext_vector_type(16))) _Float16 v16h;
typedef __attribute__((ext_vector_type(8)))  float    v8f;

constexpr int DMODEL = 768;
constexpr int NHEAD  = 12;
constexpr int HDIM   = 64;
constexpr int FFDIM  = 3072;
constexpr int NTOK   = 12288;   // 8*512 + 8*1024
constexpr int NTOK1  = 4096;    // tokens belonging to x1
constexpr int KF_D   = DMODEL / 32;  // 24
constexpr int KF_FF  = FFDIM  / 32;  // 96

// ---- fragment index helpers (CDNA5 ISA 7.12.2, 16-bit 16x32 A / 32x16 B) ----
// Per lane: 16 halves. lanes 0-15 hold K={0..7,16..23}, lanes 16-31 hold K={8..15,24..31}.
__device__ __forceinline__ void fragpos(int kl, int& half, int& e) {
  half = (kl >> 3) & 1;
  e = (kl & 7) + (kl >= 16 ? 8 : 0);
}

__device__ __forceinline__ v16h ldfrag_g(const _Float16* p) {
  return *(const v16h*)p;   // 32B aligned -> 2x global_load_b128
}

__device__ __forceinline__ v8f zero8() {
  v8f z = {0.f, 0.f, 0.f, 0.f, 0.f, 0.f, 0.f, 0.f};
  return z;
}

__device__ __forceinline__ float rmax16(float v) {
  v = fmaxf(v, __shfl_xor(v, 1, 32));
  v = fmaxf(v, __shfl_xor(v, 2, 32));
  v = fmaxf(v, __shfl_xor(v, 4, 32));
  v = fmaxf(v, __shfl_xor(v, 8, 32));
  return v;
}
__device__ __forceinline__ float rsum16(float v) {
  v += __shfl_xor(v, 1, 32);
  v += __shfl_xor(v, 2, 32);
  v += __shfl_xor(v, 4, 32);
  v += __shfl_xor(v, 8, 32);
  return v;
}

// ---- weight fp32 -> f16 B-fragment pack: Wpack[((kt*Nf + nt)*32 + lane)*16 + e]
__global__ void pack_w_kernel(const float* __restrict__ w, _Float16* __restrict__ dst,
                              int K, int N) {
  int idx = blockIdx.x * 256 + threadIdx.x;
  if (idx >= K * N) return;
  int k = idx / N, n = idx % N;
  int half, e; fragpos(k & 31, half, e);
  int lane = (n & 15) + 16 * half;
  size_t o = ((size_t)((k >> 5) * (N >> 4) + (n >> 4)) * 32 + lane) * 16 + e;
  dst[o] = (_Float16)w[(size_t)k * N + n];
}

// ---- LayerNorm -> f16 A-fragment pack (K = DMODEL) ----
__global__ void ln_pack_kernel(const float* __restrict__ x0, const float* __restrict__ x1,
                               int split, const float* __restrict__ g,
                               const float* __restrict__ bia, _Float16* __restrict__ dst) {
  int t = blockIdx.x, tid = threadIdx.x;
  const float* xp = (t < split) ? (x0 + (size_t)t * DMODEL)
                                : (x1 + (size_t)(t - split) * DMODEL);
  __shared__ float ss[256], sq[256];
  float a = 0.f, b = 0.f;
  for (int k = tid; k < DMODEL; k += 256) { float v = xp[k]; a += v; b += v * v; }
  ss[tid] = a; sq[tid] = b; __syncthreads();
  for (int off = 128; off > 0; off >>= 1) {
    if (tid < off) { ss[tid] += ss[tid + off]; sq[tid] += sq[tid + off]; }
    __syncthreads();
  }
  float mean = ss[0] * (1.0f / DMODEL);
  float var  = sq[0] * (1.0f / DMODEL) - mean * mean;
  float rs = rsqrtf(var + 1e-5f);
  int mt = t >> 4, rl = t & 15;
  for (int k = tid; k < DMODEL; k += 256) {
    float v = (xp[k] - mean) * rs * g[k] + bia[k];
    int half, e; fragpos(k & 31, half, e);
    int lane = rl + 16 * half;
    size_t o = ((size_t)(mt * KF_D + (k >> 5)) * 32 + lane) * 16 + e;
    dst[o] = (_Float16)v;
  }
}

// ---- f16 row-major [NTOK,K] -> A-fragment pack ----
__global__ void pack_a_kernel(const _Float16* __restrict__ src,
                              _Float16* __restrict__ dst, int K) {
  long long idx = (long long)blockIdx.x * 256 + threadIdx.x;
  if (idx >= (long long)NTOK * K) return;
  int r = (int)(idx / K), k = (int)(idx % K);
  int half, e; fragpos(k & 31, half, e);
  int lane = (r & 15) + 16 * half;
  size_t o = ((size_t)((r >> 4) * (K >> 5) + (k >> 5)) * 32 + lane) * 16 + e;
  dst[o] = src[idx];
}

// ---- qkv f16 [NTOK, 3*D] -> attention fragment layouts ----
// Qpack: per (qtile,head): 2 A-frags (hd 0..31 / 32..63), scaled by 1/8.
// Kpack: per (16-key tile,head): 2 B-frags (n=key local, k=hd chunk).
// Vpack: per (32-key tile,head): 4 B-frags (n=hd col group, k=key local).
__global__ void pack_qkv_kernel(const _Float16* __restrict__ qkv,
                                _Float16* __restrict__ qp,
                                _Float16* __restrict__ kp,
                                _Float16* __restrict__ vp) {
  long long idx = (long long)blockIdx.x * 256 + threadIdx.x;
  if (idx >= (long long)NTOK * (3 * DMODEL)) return;
  int t = (int)(idx / (3 * DMODEL)), c = (int)(idx % (3 * DMODEL));
  _Float16 val = qkv[idx];
  int which = c / DMODEL;
  int cd = c % DMODEL;
  int h = cd >> 6, d = cd & 63;
  if (which == 0) {
    float f = (float)val * 0.125f;          // 1/sqrt(64)
    int half, e; fragpos(d & 31, half, e);
    int lane = (t & 15) + 16 * half;
    size_t o = ((size_t)((t >> 4) * NHEAD + h) * 2 + (d >> 5)) * 512 + lane * 16 + e;
    qp[o] = (_Float16)f;
  } else if (which == 1) {
    int half, e; fragpos(d & 31, half, e);
    int lane = (t & 15) + 16 * half;
    size_t o = ((size_t)((t >> 4) * NHEAD + h) * 2 + (d >> 5)) * 512 + lane * 16 + e;
    kp[o] = val;
  } else {
    int half, e; fragpos(t & 31, half, e);
    int lane = (d & 15) + 16 * half;
    size_t o = ((size_t)((t >> 5) * NHEAD + h) * 4 + (d >> 4)) * 512 + lane * 16 + e;
    vp[o] = val;
  }
}

// ---- flash attention: one wave per (16-query tile, head) ----
__global__ void __launch_bounds__(32) attn_kernel(const _Float16* __restrict__ qp,
                                                  const _Float16* __restrict__ kp,
                                                  const _Float16* __restrict__ vp,
                                                  _Float16* __restrict__ o16) {
  int bid = blockIdx.x;
  int qt = bid / NHEAD, h = bid % NHEAD;
  int lane = threadIdx.x;
  int half = lane >> 4, l15 = lane & 15;
  int t0 = qt * 16;
  int seq_start, seq_len;
  if (t0 < NTOK1) { seq_start = t0 & ~511;                  seq_len = 512; }
  else            { seq_start = NTOK1 + ((t0 - NTOK1) & ~1023); seq_len = 1024; }

  v16h qf0 = ldfrag_g(qp + ((size_t)(qt * NHEAD + h) * 2 + 0) * 512 + lane * 16);
  v16h qf1 = ldfrag_g(qp + ((size_t)(qt * NHEAD + h) * 2 + 1) * 512 + lane * 16);

  v8f of[4];
  float m[8], l[8];
#pragma unroll
  for (int j = 0; j < 4; j++) of[j] = zero8();
#pragma unroll
  for (int v = 0; v < 8; v++) { m[v] = -1e30f; l[v] = 0.0f; }

  __shared__ _Float16 sP[16 * 32];     // 1 KB transpose tile (C-layout -> A-layout)

  int nit = seq_len >> 5;
  int kt16 = seq_start >> 4;
  int kt32 = seq_start >> 5;
  for (int it = 0; it < nit; ++it) {
    int j0 = kt16 + it * 2, j1 = j0 + 1;
    v8f s0 = zero8(), s1 = zero8();
    v16h kf;
    kf = ldfrag_g(kp + ((size_t)(j0 * NHEAD + h) * 2 + 0) * 512 + lane * 16);
    s0 = __builtin_amdgcn_wmma_f32_16x16x32_f16(false, qf0, false, kf, (short)0, s0, false, false);
    kf = ldfrag_g(kp + ((size_t)(j0 * NHEAD + h) * 2 + 1) * 512 + lane * 16);
    s0 = __builtin_amdgcn_wmma_f32_16x16x32_f16(false, qf1, false, kf, (short)0, s0, false, false);
    kf = ldfrag_g(kp + ((size_t)(j1 * NHEAD + h) * 2 + 0) * 512 + lane * 16);
    s1 = __builtin_amdgcn_wmma_f32_16x16x32_f16(false, qf0, false, kf, (short)0, s1, false, false);
    kf = ldfrag_g(kp + ((size_t)(j1 * NHEAD + h) * 2 + 1) * 512 + lane * 16);
    s1 = __builtin_amdgcn_wmma_f32_16x16x32_f16(false, qf1, false, kf, (short)0, s1, false, false);

#pragma unroll
    for (int v = 0; v < 8; v++) {
      float mx = rmax16(fmaxf(s0[v], s1[v]));
      float mn = fmaxf(m[v], mx);
      float sc = __expf(m[v] - mn);
      float p0 = __expf(s0[v] - mn);
      float p1 = __expf(s1[v] - mn);
      float rs = rsum16(p0 + p1);
      l[v] = l[v] * sc + rs;
      m[v] = mn;
#pragma unroll
      for (int j = 0; j < 4; j++) of[j][v] *= sc;
      int row = v + 8 * half;
      sP[row * 32 + l15]      = (_Float16)p0;   // keys 0..15 of this 32-tile
      sP[row * 32 + 16 + l15] = (_Float16)p1;   // keys 16..31
    }
    __syncthreads();   // single-wave block: lowers to s_nop, DS in-order anyway

    v16h pf;
    {
      const _Float16* pb = &sP[l15 * 32 + 8 * half];
      ((int4*)&pf)[0] = *(const int4*)pb;         // k = 8h..8h+7
      ((int4*)&pf)[1] = *(const int4*)(pb + 16);  // k = 16+8h..16+8h+7
    }
#pragma unroll
    for (int j = 0; j < 4; j++) {
      v16h vf = ldfrag_g(vp + ((size_t)((kt32 + it) * NHEAD + h) * 4 + j) * 512 + lane * 16);
      of[j] = __builtin_amdgcn_wmma_f32_16x16x32_f16(false, pf, false, vf, (short)0, of[j], false, false);
    }
  }

#pragma unroll
  for (int v = 0; v < 8; v++) {
    float inv = 1.0f / l[v];
    int row = t0 + v + 8 * half;
#pragma unroll
    for (int j = 0; j < 4; j++)
      o16[(size_t)row * DMODEL + h * HDIM + j * 16 + l15] = (_Float16)(of[j][v] * inv);
  }
}

// ---- WMMA GEMM: C[M,N] = Apack @ Wpack, 128x128 block tile, 32x64 wave tile ----
enum { EPI_F16 = 0, EPI_GELU_F16 = 1, EPI_RESID_F32 = 2 };

template <int EPI>
__global__ void __launch_bounds__(256) gemm_wmma_kernel(
    const _Float16* __restrict__ Apack, const _Float16* __restrict__ Wpack,
    const float* __restrict__ bias, int N, int Kfrags,
    _Float16* __restrict__ out16, float* __restrict__ out32,
    const float* __restrict__ ls, const float* __restrict__ resid0,
    const float* __restrict__ resid1, int split) {
  int w = threadIdx.x >> 5, lane = threadIdx.x & 31;
  int wm = w >> 1, wn = w & 1;
  int Nfrags = N >> 4;
  int mbase = blockIdx.x * 128 + wm * 32;
  int nbase = blockIdx.y * 128 + wn * 64;
  int mt0 = mbase >> 4, nt0 = nbase >> 4;

  v8f acc[2][4];
#pragma unroll
  for (int i = 0; i < 2; i++)
#pragma unroll
    for (int j = 0; j < 4; j++) acc[i][j] = zero8();

  for (int kt = 0; kt < Kfrags; ++kt) {
    v16h a[2], b[4];
#pragma unroll
    for (int i = 0; i < 2; i++)
      a[i] = ldfrag_g(Apack + ((size_t)(mt0 + i) * Kfrags + kt) * 512 + lane * 16);
#pragma unroll
    for (int j = 0; j < 4; j++)
      b[j] = ldfrag_g(Wpack + ((size_t)kt * Nfrags + (nt0 + j)) * 512 + lane * 16);
    if (kt + 1 < Kfrags)   // global_prefetch_b8 for next B panel
      __builtin_prefetch(Wpack + ((size_t)(kt + 1) * Nfrags + nt0) * 512 + lane * 16, 0, 3);
#pragma unroll
    for (int i = 0; i < 2; i++)
#pragma unroll
      for (int j = 0; j < 4; j++)
        acc[i][j] = __builtin_amdgcn_wmma_f32_16x16x32_f16(false, a[i], false, b[j],
                                                           (short)0, acc[i][j], false, false);
  }

  int half = lane >> 4, l15 = lane & 15;
#pragma unroll
  for (int i = 0; i < 2; i++) {
#pragma unroll
    for (int j = 0; j < 4; j++) {
      int col = nbase + j * 16 + l15;
      float bi = bias[col];
#pragma unroll
      for (int v = 0; v < 8; v++) {
        int row = mbase + i * 16 + v + 8 * half;
        float val = acc[i][j][v] + bi;
        if (EPI == EPI_F16) {
          out16[(size_t)row * N + col] = (_Float16)val;
        } else if (EPI == EPI_GELU_F16) {
          float g = 0.5f * val * (1.0f + erff(val * 0.70710678118f));
          out16[(size_t)row * N + col] = (_Float16)g;
        } else {
          const float* rp = (row < split) ? (resid0 + (size_t)row * N)
                                          : (resid1 + (size_t)(row - split) * N);
          out32[(size_t)row * N + col] = rp[col] + ls[col] * val;
        }
      }
    }
  }
}

extern "C" void kernel_launch(void* const* d_in, const int* in_sizes, int n_in,
                              void* d_out, int out_size, void* d_ws, size_t ws_size,
                              hipStream_t stream) {
  const float* x1     = (const float*)d_in[0];
  const float* x2     = (const float*)d_in[1];
  const float* ln1_g  = (const float*)d_in[2];
  const float* ln1_b  = (const float*)d_in[3];
  const float* w_qkv  = (const float*)d_in[4];
  const float* b_qkv  = (const float*)d_in[5];
  const float* w_proj = (const float*)d_in[6];
  const float* b_proj = (const float*)d_in[7];
  const float* ls1    = (const float*)d_in[8];
  const float* ln2_g  = (const float*)d_in[9];
  const float* ln2_b  = (const float*)d_in[10];
  const float* w_fc1  = (const float*)d_in[11];
  const float* b_fc1  = (const float*)d_in[12];
  const float* w_fc2  = (const float*)d_in[13];
  const float* b_fc2  = (const float*)d_in[14];
  const float* ls2    = (const float*)d_in[15];
  float* out = (float*)d_out;

  // workspace carve (regions reused stream-ordered; ~260 MB total)
  char* p = (char*)d_ws;
  _Float16* Wq = (_Float16*)p;    p += (size_t)DMODEL * 3 * DMODEL * 2;
  _Float16* Wp = (_Float16*)p;    p += (size_t)DMODEL * DMODEL * 2;
  _Float16* W1 = (_Float16*)p;    p += (size_t)DMODEL * FFDIM * 2;
  _Float16* W2 = (_Float16*)p;    p += (size_t)FFDIM * DMODEL * 2;
  _Float16* hpack = (_Float16*)p; p += (size_t)NTOK * FFDIM * 2;  // A-frag buffer (max K=FF)
  _Float16* buf16 = (_Float16*)p; p += (size_t)NTOK * FFDIM * 2;  // qkv / o / gelu f16 rows
  _Float16* qp = (_Float16*)p;    p += (size_t)NTOK * DMODEL * 2;
  _Float16* kp = (_Float16*)p;    p += (size_t)NTOK * DMODEL * 2;
  _Float16* vp = (_Float16*)p;    p += (size_t)NTOK * DMODEL * 2;
  float* x2sc = (float*)p;        p += (size_t)NTOK * DMODEL * 4;

  pack_w_kernel<<<(DMODEL * 3 * DMODEL + 255) / 256, 256, 0, stream>>>(w_qkv, Wq, DMODEL, 3 * DMODEL);
  pack_w_kernel<<<(DMODEL * DMODEL + 255) / 256, 256, 0, stream>>>(w_proj, Wp, DMODEL, DMODEL);
  pack_w_kernel<<<(DMODEL * FFDIM + 255) / 256, 256, 0, stream>>>(w_fc1, W1, DMODEL, FFDIM);
  pack_w_kernel<<<(FFDIM * DMODEL + 255) / 256, 256, 0, stream>>>(w_fc2, W2, FFDIM, DMODEL);

  // LN1 -> hpack;  QKV GEMM -> buf16 (f16 rows)
  ln_pack_kernel<<<NTOK, 256, 0, stream>>>(x1, x2, NTOK1, ln1_g, ln1_b, hpack);
  gemm_wmma_kernel<EPI_F16><<<dim3(NTOK / 128, (3 * DMODEL) / 128), 256, 0, stream>>>(
      hpack, Wq, b_qkv, 3 * DMODEL, KF_D, buf16, nullptr, nullptr, nullptr, nullptr, 0);

  // repack qkv; attention -> buf16 (o rows, qkv region dead)
  pack_qkv_kernel<<<(int)(((long long)NTOK * 3 * DMODEL + 255) / 256), 256, 0, stream>>>(buf16, qp, kp, vp);
  attn_kernel<<<(NTOK / 16) * NHEAD, 32, 0, stream>>>(qp, kp, vp, buf16);

  // proj GEMM + LayerScale + residual -> x2sc
  pack_a_kernel<<<(int)(((long long)NTOK * DMODEL + 255) / 256), 256, 0, stream>>>(buf16, hpack, DMODEL);
  gemm_wmma_kernel<EPI_RESID_F32><<<dim3(NTOK / 128, DMODEL / 128), 256, 0, stream>>>(
      hpack, Wp, b_proj, DMODEL, KF_D, nullptr, x2sc, ls1, x1, x2, NTOK1);

  // LN2 -> hpack;  FC1 + GELU -> buf16;  repack;  FC2 + LayerScale + residual -> out
  ln_pack_kernel<<<NTOK, 256, 0, stream>>>(x2sc, x2sc, 0, ln2_g, ln2_b, hpack);
  gemm_wmma_kernel<EPI_GELU_F16><<<dim3(NTOK / 128, FFDIM / 128), 256, 0, stream>>>(
      hpack, W1, b_fc1, FFDIM, KF_D, buf16, nullptr, nullptr, nullptr, nullptr, 0);
  pack_a_kernel<<<(int)(((long long)NTOK * FFDIM + 255) / 256), 256, 0, stream>>>(buf16, hpack, FFDIM);
  gemm_wmma_kernel<EPI_RESID_F32><<<dim3(NTOK / 128, DMODEL / 128), 256, 0, stream>>>(
      hpack, W2, b_fc2, DMODEL, KF_FF, nullptr, out, ls2, x2sc, x2sc, 0);
}
